// EncoderBlock_73083163509540
// MI455X (gfx1250) — matrix operand order repaired and checked by reading, not compiled, not run on hardware
//
#include <hip/hip_runtime.h>
#include <math.h>

// ---------------------------------------------------------------------------
// Types for CDNA5 WMMA (wave32): v_wmma_f32_16x16x32_bf16
// ---------------------------------------------------------------------------
typedef __attribute__((ext_vector_type(16))) __bf16 bf16x16;
typedef __attribute__((ext_vector_type(8)))  __bf16 bf16x8;
typedef __attribute__((ext_vector_type(8)))  float  f32x8;

__device__ __forceinline__ f32x8 wmma_bf16(bf16x16 a, bf16x16 b, f32x8 c) {
  // (neg_a, A, neg_b, B, c_mod, C, reuse_a, reuse_b)
  return __builtin_amdgcn_wmma_f32_16x16x32_bf16(false, a, false, b, (short)0, c,
                                                 false, false);
}

// B-fragment: 16 contiguous bf16 per lane (k = half*16 + e)
__device__ __forceinline__ bf16x16 ldfrag_b(const __bf16* p) {
  bf16x8 lo = *(const bf16x8*)p;
  bf16x8 hi = *(const bf16x8*)(p + 8);
  return __builtin_shufflevector(lo, hi, 0,1,2,3,4,5,6,7,8,9,10,11,12,13,14,15);
}
// A-fragment: 16 contiguous bf16 per lane (k = half*16 + e)
__device__ __forceinline__ bf16x16 ldfrag_a(const __bf16* p) {
  bf16x8 lo = *(const bf16x8*)p;
  bf16x8 hi = *(const bf16x8*)(p + 8);
  return __builtin_shufflevector(lo, hi, 0,1,2,3,4,5,6,7,8,9,10,11,12,13,14,15);
}

// ---------------------------------------------------------------------------
// Constants
// ---------------------------------------------------------------------------
#define BB 4
#define NN 1024
#define CC 768
#define HH 12
#define DD 64
#define MM 3072
#define RR (BB * NN)      // 4096 token rows
#define QKVC (3 * CC)     // 2304

// ---------------------------------------------------------------------------
// Kernel: f32 -> bf16 cast (4 floats / thread)
// ---------------------------------------------------------------------------
__global__ __launch_bounds__(256) void k_cast_bf16(const float* __restrict__ in,
                                                   __bf16* __restrict__ out, int n4) {
  int i = blockIdx.x * 256 + threadIdx.x;
  if (i < n4) {
    float4 v = ((const float4*)in)[i];
    out[i * 4 + 0] = (__bf16)v.x;
    out[i * 4 + 1] = (__bf16)v.y;
    out[i * 4 + 2] = (__bf16)v.z;
    out[i * 4 + 3] = (__bf16)v.w;
  }
}

// ---------------------------------------------------------------------------
// Kernel: LayerNorm over C=768, one block (256 thr) per row.
// WRITE_F32: also write f32 copy (for residual use).
// ---------------------------------------------------------------------------
template <bool WRITE_F32>
__global__ __launch_bounds__(256) void k_layernorm(const float* __restrict__ x,
                                                   const float* __restrict__ g,
                                                   const float* __restrict__ be,
                                                   float* __restrict__ out_f32,
                                                   __bf16* __restrict__ out_bf16) {
  __shared__ float red[256];
  const int row = blockIdx.x, tid = threadIdx.x;
  const float* xr = x + (size_t)row * CC;
  float v[3];
  float s = 0.f;
#pragma unroll
  for (int j = 0; j < 3; ++j) { v[j] = xr[tid + j * 256]; s += v[j]; }
  red[tid] = s; __syncthreads();
  for (int off = 128; off > 0; off >>= 1) {
    if (tid < off) red[tid] += red[tid + off];
    __syncthreads();
  }
  const float mu = red[0] * (1.0f / CC);
  __syncthreads();
  float s2 = 0.f;
#pragma unroll
  for (int j = 0; j < 3; ++j) { float d = v[j] - mu; s2 += d * d; }
  red[tid] = s2; __syncthreads();
  for (int off = 128; off > 0; off >>= 1) {
    if (tid < off) red[tid] += red[tid + off];
    __syncthreads();
  }
  const float rstd = rsqrtf(red[0] * (1.0f / CC) + 1e-5f);
#pragma unroll
  for (int j = 0; j < 3; ++j) {
    const int c = tid + j * 256;
    const float y = (v[j] - mu) * rstd * g[c] + be[c];
    if (WRITE_F32) out_f32[(size_t)row * CC + c] = y;
    out_bf16[(size_t)row * CC + c] = (__bf16)y;
  }
}

// ---------------------------------------------------------------------------
// Kernel: bf16 WMMA GEMM  out[M,N] = A[M,K] * B[K,N]  (+bias, +res, gelu)
// Block tile 128x256, BK=32, 256 threads = 8 waves (2x4), wave tile 64x64.
// 16 WMMAs per 16 ds_load_b128 per wave per k-step (4x fragment reuse).
// Software-pipelined: next K-slice is global-loaded into registers while the
// current LDS tile feeds the WMMAs. Epilogue fully compile-time specialized.
// ---------------------------------------------------------------------------
template <bool BIAS, bool GELU_F, bool RES_A, bool RES_B, bool OUT_F, bool OUT_H>
__global__ __launch_bounds__(256) void k_gemm(
    const __bf16* __restrict__ A, const __bf16* __restrict__ Bm,
    int M, int N, int K,
    const float* __restrict__ bias,
    const float* __restrict__ resA, const float* __restrict__ resB,
    float* __restrict__ outF, __bf16* __restrict__ outH) {
  __shared__ __align__(16) __bf16 aS[128][40];   // [m][k], padded
  __shared__ __align__(16) __bf16 bS[256][40];   // [n][k] (transposed), padded

  const int tid = threadIdx.x;
  const int lane = tid & 31, wid = tid >> 5;
  const int hl = lane >> 4, l16 = lane & 15;
  const int m0 = blockIdx.y * 128, n0 = blockIdx.x * 256;
  const int wm0 = (wid & 1) * 64, wn0 = (wid >> 1) * 64;

  // cooperative-load geometry (per thread)
  const int ar = tid >> 1, acb = (tid & 1) * 16;          // A: row, col-16-block
  const int bkr = tid >> 3, bnb = (tid & 7) * 32;         // B: k-row, n-32-block
  const __bf16* aSrc = A + (size_t)(m0 + ar) * K + acb;
  const __bf16* bSrc = Bm + (size_t)bkr * N + n0 + bnb;

  f32x8 acc[4][4] = {};

  uint4 ra0, ra1, rb[4];
  // prologue: fetch slice 0
  ra0 = *(const uint4*)aSrc;
  ra1 = *(const uint4*)(aSrc + 8);
#pragma unroll
  for (int i = 0; i < 4; ++i) rb[i] = ((const uint4*)bSrc)[i];

  const int nk = K / 32;
  for (int ks = 0; ks < nk; ++ks) {
    // commit registers -> LDS
    *(uint4*)&aS[ar][acb] = ra0;
    *(uint4*)&aS[ar][acb + 8] = ra1;
    {
      union { uint4 u[4]; __bf16 e[32]; } t;
#pragma unroll
      for (int i = 0; i < 4; ++i) t.u[i] = rb[i];
#pragma unroll
      for (int i = 0; i < 32; ++i) bS[bnb + i][bkr] = t.e[i];
    }
    __syncthreads();

    // prefetch next slice while WMMAs run
    if (ks + 1 < nk) {
      const __bf16* an = aSrc + (ks + 1) * 32;
      const __bf16* bn = bSrc + (size_t)(ks + 1) * 32 * N;
      ra0 = *(const uint4*)an;
      ra1 = *(const uint4*)(an + 8);
#pragma unroll
      for (int i = 0; i < 4; ++i) rb[i] = ((const uint4*)bn)[i];
    }

    bf16x16 bf[4], af[4];
#pragma unroll
    for (int nt = 0; nt < 4; ++nt)
      bf[nt] = ldfrag_b(&bS[wn0 + nt * 16 + l16][hl * 16]);
#pragma unroll
    for (int mt = 0; mt < 4; ++mt)
      af[mt] = ldfrag_a(&aS[wm0 + mt * 16 + l16][hl * 16]);
#pragma unroll
    for (int mt = 0; mt < 4; ++mt)
#pragma unroll
      for (int nt = 0; nt < 4; ++nt)
        acc[mt][nt] = wmma_bf16(af[mt], bf[nt], acc[mt][nt]);
    __syncthreads();
  }

  // Epilogue. C/D layout: elem j -> row m = j + 8*hl (within 16), col n = l16.
#pragma unroll
  for (int mt = 0; mt < 4; ++mt) {
#pragma unroll
    for (int nt = 0; nt < 4; ++nt) {
      const int n = n0 + wn0 + nt * 16 + l16;
      const int mb = m0 + wm0 + mt * 16 + hl * 8;
      float bv = 0.f;
      if (BIAS) bv = bias[n];
      size_t idx = (size_t)mb * N + n;
#pragma unroll
      for (int j = 0; j < 8; ++j, idx += N) {
        float v = acc[mt][nt][j] + bv;
        if (RES_A) v += resA[idx];
        if (RES_B) v += resB[idx];
        if (GELU_F) v = 0.5f * v * (1.0f + erff(v * 0.70710678118654752f));
        if (OUT_F) outF[idx] = v;
        if (OUT_H) outH[idx] = (__bf16)v;
      }
    }
  }
}

// ---------------------------------------------------------------------------
// Kernel: fused flash attention for one (b, h, 128-query tile) per block.
// qkv layout: [B*N, 3*C], q at col h*64, k at 768+h*64, v at 1536+h*64.
// Each wave owns 16 queries. K chunk (128x64) and V^T chunk (64x128) in LDS.
// Scores computed transposed (S^T = K * Q^T) so softmax is lane-local in q.
// attn^T = V^T * P^T accumulated with WMMA; P^T staged via per-wave LDS.
// ---------------------------------------------------------------------------
__global__ __launch_bounds__(256) void k_attention(const __bf16* __restrict__ qkv,
                                                   __bf16* __restrict__ attn_out) {
  __shared__ __align__(16) __bf16 kS[128][72];    // [key][d]
  __shared__ __align__(16) __bf16 vtS[64][136];   // [d][key]
  __shared__ __align__(16) __bf16 pS[8][16][32];  // [wave][q][key]  (P^T)

  const int tid = threadIdx.x;
  const int lane = tid & 31, wid = tid >> 5;
  const int hl = lane >> 4, l16 = lane & 15;
  const int qt = blockIdx.x, h = blockIdx.y, b = blockIdx.z;

  const __bf16* qkv_b = qkv + (size_t)b * NN * QKVC;
  const int q = qt * 128 + wid * 16 + l16;  // this lane's query (dup per half)

  // Q^T B-fragments (k-dim = d): frag0 d0..31, frag1 d32..63
  const __bf16* qrow = qkv_b + (size_t)q * QKVC + h * DD;
  const bf16x16 bq0 = ldfrag_b(qrow + hl * 16);
  const bf16x16 bq1 = ldfrag_b(qrow + 32 + hl * 16);

  float m_run = -3.0e38f, l_run = 0.f;
  f32x8 acc[4] = {};
  const float scale = 0.125f;  // 1/sqrt(64)

  for (int c = 0; c < 8; ++c) {        // 8 chunks of 128 keys
    const int kb = c * 128;
    { // cooperative K / V^T load: thread -> (key = tid/2, 32 d's)
      const int key = tid >> 1, db = (tid & 1) * 32;
      const __bf16* ksrc = qkv_b + (size_t)(kb + key) * QKVC + CC + h * DD + db;
      const __bf16* vsrc = qkv_b + (size_t)(kb + key) * QKVC + 2 * CC + h * DD + db;
      union { uint4 u[4]; __bf16 e[32]; } kk, vv;
#pragma unroll
      for (int i = 0; i < 4; ++i) kk.u[i] = ((const uint4*)ksrc)[i];
#pragma unroll
      for (int i = 0; i < 4; ++i) vv.u[i] = ((const uint4*)vsrc)[i];
#pragma unroll
      for (int i = 0; i < 4; ++i) *(uint4*)&kS[key][db + i * 8] = kk.u[i];
#pragma unroll
      for (int i = 0; i < 32; ++i) vtS[db + i][key] = vv.e[i];
    }
    __syncthreads();

    for (int g = 0; g < 4; ++g) {      // 32-key groups
      const int kg = g * 32;
      f32x8 st[2];
#pragma unroll
      for (int s = 0; s < 2; ++s) {    // two 16-key S^T tiles
        bf16x16 ak0 = ldfrag_a(&kS[kg + s * 16 + l16][hl * 16]);
        bf16x16 ak1 = ldfrag_a(&kS[kg + s * 16 + l16][32 + hl * 16]);
        f32x8 z = {};
        z = wmma_bf16(ak0, bq0, z);
        z = wmma_bf16(ak1, bq1, z);
        st[s] = z * scale;
      }
      // online softmax over this lane's q (keys split across wave halves)
      float tmax = -3.0e38f;
#pragma unroll
      for (int s = 0; s < 2; ++s)
#pragma unroll
        for (int j = 0; j < 8; ++j) tmax = fmaxf(tmax, st[s][j]);
      tmax = fmaxf(tmax, __shfl_xor(tmax, 16, 32));
      const float m_new = fmaxf(m_run, tmax);
      const float alpha = __expf(m_run - m_new);
      float psum = 0.f;
#pragma unroll
      for (int s = 0; s < 2; ++s)
#pragma unroll
        for (int j = 0; j < 8; ++j) {
          const float p = __expf(st[s][j] - m_new);
          psum += p;
          pS[wid][l16][s * 16 + j + hl * 8] = (__bf16)p;  // P^T[key][q]
        }
      psum += __shfl_xor(psum, 16, 32);
      l_run = l_run * alpha + psum;
      m_run = m_new;
#pragma unroll
      for (int t = 0; t < 4; ++t) acc[t] = acc[t] * alpha;

      // attn^T += V^T(16d x 32key) * P^T(32key x 16q)
      const bf16x16 bp = ldfrag_b(&pS[wid][l16][hl * 16]);
#pragma unroll
      for (int t = 0; t < 4; ++t) {
        bf16x16 av = ldfrag_a(&vtS[t * 16 + l16][kg + hl * 16]);
        acc[t] = wmma_bf16(av, bp, acc[t]);
      }
    }
    __syncthreads();
  }

  const float inv_l = 1.0f / l_run;
  __bf16* orow = attn_out + ((size_t)b * NN + q) * CC + h * DD;
#pragma unroll
  for (int t = 0; t < 4; ++t)
#pragma unroll
    for (int j = 0; j < 8; ++j) {
      const int d = t * 16 + j + hl * 8;
      orow[d] = (__bf16)(acc[t][j] * inv_l);
    }
}

// ---------------------------------------------------------------------------
// Launcher
// ---------------------------------------------------------------------------
extern "C" void kernel_launch(void* const* d_in, const int* in_sizes, int n_in,
                              void* d_out, int out_size, void* d_ws, size_t ws_size,
                              hipStream_t stream) {
  (void)in_sizes; (void)n_in; (void)out_size; (void)ws_size;
  const float* x     = (const float*)d_in[0];
  const float* ln1_g = (const float*)d_in[1];
  const float* ln1_b = (const float*)d_in[2];
  const float* w_qkv = (const float*)d_in[3];
  const float* w_out = (const float*)d_in[4];
  const float* b_out = (const float*)d_in[5];
  const float* ln2_g = (const float*)d_in[6];
  const float* ln2_b = (const float*)d_in[7];
  const float* w1    = (const float*)d_in[8];
  const float* b1    = (const float*)d_in[9];
  const float* w2    = (const float*)d_in[10];
  const float* b2    = (const float*)d_in[11];

  char* ws = (char*)d_ws;
  size_t off = 0;
  auto carve = [&](size_t bytes) { char* p = ws + off; off += (bytes + 255) & ~(size_t)255; return p; };
  __bf16* wqkv_h = (__bf16*)carve((size_t)CC * QKVC * 2);
  __bf16* wout_h = (__bf16*)carve((size_t)CC * CC * 2);
  __bf16* w1_h   = (__bf16*)carve((size_t)CC * MM * 2);
  __bf16* w2_h   = (__bf16*)carve((size_t)MM * CC * 2);
  float*  h_f    = (float*)carve((size_t)RR * CC * 4);
  __bf16* h_h    = (__bf16*)carve((size_t)RR * CC * 2);
  __bf16* qkv_h  = (__bf16*)carve((size_t)RR * QKVC * 2);
  __bf16* attn_h = (__bf16*)carve((size_t)RR * CC * 2);
  float*  y1_f   = (float*)carve((size_t)RR * CC * 4);
  __bf16* h2_h   = (__bf16*)carve((size_t)RR * CC * 2);
  __bf16* g_h    = (__bf16*)carve((size_t)RR * MM * 2);

  // 1) cast weights to bf16
  k_cast_bf16<<<(CC * QKVC) / 1024, 256, 0, stream>>>(w_qkv, wqkv_h, (CC * QKVC) / 4);
  k_cast_bf16<<<(CC * CC) / 1024, 256, 0, stream>>>(w_out, wout_h, (CC * CC) / 4);
  k_cast_bf16<<<(CC * MM) / 1024, 256, 0, stream>>>(w1, w1_h, (CC * MM) / 4);
  k_cast_bf16<<<(MM * CC) / 1024, 256, 0, stream>>>(w2, w2_h, (MM * CC) / 4);

  // 2) LN1 -> h (f32 for residual, bf16 for GEMM)
  k_layernorm<true><<<RR, 256, 0, stream>>>(x, ln1_g, ln1_b, h_f, h_h);

  // 3) qkv = h @ w_qkv   [4096 x 2304]
  k_gemm<false, false, false, false, false, true>
      <<<dim3(QKVC / 256, RR / 128), 256, 0, stream>>>(
          h_h, wqkv_h, RR, QKVC, CC, nullptr, nullptr, nullptr, nullptr, qkv_h);

  // 4) attention -> attn_h [4096 x 768]
  k_attention<<<dim3(NN / 128, HH, BB), 256, 0, stream>>>(qkv_h, attn_h);

  // 5) y1 = x + h + attn @ w_out + b_out
  k_gemm<true, false, true, true, true, false>
      <<<dim3(CC / 256, RR / 128), 256, 0, stream>>>(
          attn_h, wout_h, RR, CC, CC, b_out, h_f, x, y1_f, nullptr);

  // 6) LN2 -> h2 (bf16 only)
  k_layernorm<false><<<RR, 256, 0, stream>>>(y1_f, ln2_g, ln2_b, nullptr, h2_h);

  // 7) g = gelu(h2 @ w1 + b1)  [4096 x 3072]
  k_gemm<true, true, false, false, false, true>
      <<<dim3(MM / 256, RR / 128), 256, 0, stream>>>(
          h2_h, w1_h, RR, MM, CC, b1, nullptr, nullptr, nullptr, g_h);

  // 8) out = y1 + g @ w2 + b2  [4096 x 768] f32
  k_gemm<true, false, true, false, true, false>
      <<<dim3(CC / 256, RR / 128), 256, 0, stream>>>(
          g_h, w2_h, RR, CC, MM, b2, y1_f, nullptr, (float*)d_out, nullptr);
}
